// Mamba3Block_37297495998518
// MI455X (gfx1250) — matrix operand order repaired and checked by reading, not compiled, not run on hardware
//
#include <hip/hip_runtime.h>
#include <math.h>

// ---------------- problem constants ----------------
#define BB     4
#define LL     2048
#define BL     (BB*LL)        // 8192 rows
#define DM     1280
#define DST    128
#define NH     16
#define HD     80
#define DIN    1280
#define DPROJ  2832           // 2*1280 + 2*128 + 16
#define CCH    1536           // conv channels (1280 + 2*128)
#define NCHK   16             // number of L-chunks
#define LC     128            // chunk length (16*128 = 2048)

// ---------------- workspace layout (bytes) ----------------
#define OFF_XBF    ((size_t)0)                                   // x bf16       : 8192*1280*2
#define OFF_WINT   (OFF_XBF   + (size_t)BL*DM*2)                 // W_in^T bf16  : 2832*1280*2
#define OFF_ZX     (OFF_WINT  + (size_t)DPROJ*DM*2)              // zxbcdt f32   : 8192*2832*4
#define OFF_XBC    (OFF_ZX    + (size_t)BL*DPROJ*4)              // conv(xBC) f32: 8192*1536*4
#define OFF_DTP    (OFF_XBC   + (size_t)BL*CCH*4)                // softplus(dt) : 8192*16*4
#define OFF_DAV    (OFF_DTP   + (size_t)BL*NH*4)                 // exp(dt*A)    : 8192*16*4
#define OFF_S      (OFF_DAV   + (size_t)BL*NH*4)                 // chunk states : 16*4*16*80*128*4
#define OFF_HINIT  (OFF_S     + (size_t)NCHK*BB*NH*HD*DST*4)     // chunk inits
#define OFF_DECAY  (OFF_HINIT + (size_t)NCHK*BB*NH*HD*DST*4)     // chunk decay  : 16*4*16*4
#define OFF_Y      (OFF_DECAY + (size_t)4096)                    // y f32        : 8192*1280*4
#define OFF_G      (OFF_Y     + (size_t)BL*DIN*4)                // g bf16       : 8192*1280*2
#define OFF_WOUTT  (OFF_G     + (size_t)BL*DIN*2)                // W_out^T bf16 : 1280*1280*2

// ---------------- types ----------------
typedef __attribute__((ext_vector_type(16))) __bf16        v16bf;
typedef __attribute__((ext_vector_type(8)))  __bf16        v8bf;
typedef __attribute__((ext_vector_type(8)))  float         v8f;

__device__ __forceinline__ unsigned short f32_to_bf16(float f) {
  unsigned int u = __float_as_uint(f);
  unsigned int r = u + 0x7FFFu + ((u >> 16) & 1u);   // round-to-nearest-even
  return (unsigned short)(r >> 16);
}

// async global->LDS copy of 32 contiguous bytes (two b128 ops; the IOFFSET
// immediate applies to BOTH the LDS and memory address per the CDNA5 ISA).
__device__ __forceinline__ void async_copy32(unsigned lds_off, const void* gptr) {
  asm volatile("global_load_async_to_lds_b128 %0, %1, off\n\t"
               "global_load_async_to_lds_b128 %0, %1, off offset:16"
               :: "v"(lds_off), "v"((unsigned long long)gptr)
               : "memory");
}
__device__ __forceinline__ void wait_async0() {
  asm volatile("s_wait_asynccnt 0x0" ::: "memory");
}

// ---------------- casts ----------------
__global__ void k_cast_bf16(const float* __restrict__ in, unsigned short* __restrict__ out, int n) {
  int i = blockIdx.x * blockDim.x + threadIdx.x;
  if (i < n) out[i] = f32_to_bf16(in[i]);
}

// W: K x N row-major (f32) -> Wt: N x K row-major (bf16)
__global__ void k_transpose_cast_bf16(const float* __restrict__ W, unsigned short* __restrict__ Wt,
                                      int K, int N) {
  int gid = blockIdx.x * blockDim.x + threadIdx.x;
  if (gid >= N * K) return;
  int n = gid / K, k = gid - n * K;
  Wt[gid] = f32_to_bf16(W[(size_t)k * N + n]);
}

// ---------------- bf16 WMMA GEMM: C[MxN] = A[MxK] * Bt[NxK]^T ----------------
// 128x128 block tile, K-step 32, double-buffered LDS filled via async
// global->LDS (ASYNCcnt) so HBM latency overlaps the WMMA pipeline.
__global__ __launch_bounds__(256)
void k_gemm_bf16(const unsigned short* __restrict__ A,   // M x K row-major (bf16 bits)
                 const unsigned short* __restrict__ Bt,  // N x K row-major (bf16 bits)
                 float* __restrict__ C, int M, int N, int K) {
  __shared__ unsigned short As[2][128][40];   // 32 K + 8 pad (bank spread)
  __shared__ unsigned short Bs[2][128][40];

  const int tid  = threadIdx.x;
  const int lane = tid & 31;
  const int wave = tid >> 5;
  const int l16  = lane & 15;
  const int half = lane >> 4;

  const int bm = blockIdx.x * 128;
  const int bn = blockIdx.y * 128;
  const int wm = (wave & 3) * 32;    // 4 waves along M
  const int wn = (wave >> 2) * 64;   // 2 waves along N

  const v8f zacc = {0.f,0.f,0.f,0.f,0.f,0.f,0.f,0.f};
  v8f acc[2][4];
#pragma unroll
  for (int i = 0; i < 2; i++)
#pragma unroll
    for (int j = 0; j < 4; j++) acc[i][j] = zacc;

  const int r  = tid >> 1;          // 0..127 tile row
  const int kk = (tid & 1) * 16;    // 0 or 16 within K-tile

  // per-thread source pointers / LDS destinations
  const unsigned short* aptr = A  + (size_t)(bm + r) * K + kk;
  const unsigned short* bptr = Bt + (size_t)(bn + r) * K + kk;
  const bool bvalid = (bn + r) < N;  // OOB rows feed only masked-out columns
  unsigned ldsA[2], ldsB[2];
#pragma unroll
  for (int u = 0; u < 2; u++) {
    ldsA[u] = (unsigned)(size_t)&As[u][r][kk];
    ldsB[u] = (unsigned)(size_t)&Bs[u][r][kk];
  }

  // prologue: prefetch tile 0 into buffer 0
  async_copy32(ldsA[0], aptr);
  if (bvalid) async_copy32(ldsB[0], bptr);

  const int ntiles = K >> 5;
  for (int kt = 0; kt < ntiles; kt++) {
    const int cur = kt & 1;
    wait_async0();        // this wave's async writes into buf[cur] landed
    __syncthreads();      // everyone's landed; prior reads of buf[cur^1] done

    if (kt + 1 < ntiles) {  // prefetch next tile into the other buffer
      const unsigned short* an = aptr + (size_t)(kt + 1) * 32;
      async_copy32(ldsA[cur ^ 1], an);
      if (bvalid) {
        const unsigned short* bnp = bptr + (size_t)(kt + 1) * 32;
        async_copy32(ldsB[cur ^ 1], bnp);
      }
    }

    v16bf afrag[2], bfrag[4];
#pragma unroll
    for (int i = 0; i < 2; i++) {       // A 16x32: lanes0-15 K{0-7,16-23}; lanes16-31 K{8-15,24-31}
      int rr = wm + i * 16 + l16;
      union { v16bf v; v8bf h[2]; } u;
      u.h[0] = *(const v8bf*)&As[cur][rr][half * 8];
      u.h[1] = *(const v8bf*)&As[cur][rr][16 + half * 8];
      afrag[i] = u.v;
    }
#pragma unroll
    for (int j = 0; j < 4; j++) {       // B 32x16: lanes0-15 K{0-7,16-23}; lanes16-31 K{8-15,24-31}
      int rr = wn + j * 16 + l16;
      union { v16bf v; v8bf h[2]; } u;
      u.h[0] = *(const v8bf*)&Bs[cur][rr][half * 8];
      u.h[1] = *(const v8bf*)&Bs[cur][rr][16 + half * 8];
      bfrag[j] = u.v;
    }

#pragma unroll
    for (int i = 0; i < 2; i++)
#pragma unroll
      for (int j = 0; j < 4; j++)
        acc[i][j] = __builtin_amdgcn_wmma_f32_16x16x32_bf16(
            false, afrag[i], false, bfrag[j], (short)0, acc[i][j], false, false);
  }

  // C/D layout: VGPR v -> M = v + 8*(lane/16), N = lane%16
#pragma unroll
  for (int i = 0; i < 2; i++) {
#pragma unroll
    for (int j = 0; j < 4; j++) {
      int nn = bn + wn + j * 16 + l16;
      if (nn < N) {
#pragma unroll
        for (int v = 0; v < 8; v++) {
          int mm = bm + wm + i * 16 + 8 * half + v;
          C[(size_t)mm * N + nn] = acc[i][j][v];
        }
      }
    }
  }
}

// ---------------- depthwise causal conv(4) + SiLU on xBC channels ----------------
__global__ void k_conv_silu(const float* __restrict__ zx, const float* __restrict__ cw,
                            const float* __restrict__ cb, float* __restrict__ xbc) {
  int gid = blockIdx.x * blockDim.x + threadIdx.x;
  if (gid >= BL * CCH) return;
  int c   = gid % CCH;
  int row = gid / CCH;          // b*L + l
  int l   = row & (LL - 1);
  float acc = cb[c];
#pragma unroll
  for (int k = 0; k < 4; k++) {
    int ll = l - 3 + k;
    if (ll >= 0)
      acc += cw[k * CCH + c] * zx[(size_t)(row - 3 + k) * DPROJ + DIN + c];
  }
  xbc[(size_t)row * CCH + c] = acc / (1.0f + __expf(-acc));   // SiLU
}

// ---------------- dt: softplus(dt + bias), dA = exp(dt * A) ----------------
__global__ void k_dt(const float* __restrict__ zx, const float* __restrict__ dt_bias,
                     const float* __restrict__ A_log, float* __restrict__ dtp,
                     float* __restrict__ dAv) {
  int gid = blockIdx.x * blockDim.x + threadIdx.x;
  if (gid >= BL * NH) return;
  int h = gid & (NH - 1);
  int row = gid >> 4;
  float v  = zx[(size_t)row * DPROJ + 2 * DIN + 2 * DST + h] + dt_bias[h];
  float sp = (v > 20.0f) ? v : log1pf(__expf(v));
  float Ah = -__expf(A_log[h]);
  dtp[gid] = sp;
  dAv[gid] = __expf(sp * Ah);
}

// ---------------- scan pass 1: per-chunk local final state (zero init) ----------------
__global__ __launch_bounds__(128)
void k_scan_state(const float* __restrict__ xbc, const float* __restrict__ dtp,
                  const float* __restrict__ dAv, const float* __restrict__ A_log,
                  float* __restrict__ S, float* __restrict__ decay) {
  const int c = blockIdx.x, b = blockIdx.y, h = blockIdx.z;
  const int tid = threadIdx.x;
  __shared__ float Bsh[DST];
  __shared__ float sc2[2];
  float hreg[DST];
#pragma unroll
  for (int n = 0; n < DST; n++) hreg[n] = 0.f;
  float sdt = 0.f;

  for (int t = 0; t < LC; t++) {
    int row = b * LL + c * LC + t;
    Bsh[tid] = xbc[(size_t)row * CCH + DIN + tid];
    if (tid == 0) { sc2[0] = dtp[row * NH + h]; sc2[1] = dAv[row * NH + h]; }
    __syncthreads();
    float dt = sc2[0], dA = sc2[1];
    sdt += dt;
    if (tid < HD) {
      float dtx = dt * xbc[(size_t)row * CCH + h * HD + tid];
#pragma unroll
      for (int n = 0; n < DST; n++) hreg[n] = hreg[n] * dA + dtx * Bsh[n];
    }
    __syncthreads();
  }
  if (tid < HD) {
    size_t sbase = ((((size_t)c * BB + b) * NH + h) * HD + tid) * DST;
#pragma unroll
    for (int n = 0; n < DST; n++) S[sbase + n] = hreg[n];
  }
  if (tid == 0) {
    float Ah = -__expf(A_log[h]);
    decay[(c * BB + b) * NH + h] = __expf(Ah * sdt);
  }
}

// ---------------- scan pass 2: chain chunk states (16 sequential chunks) ----------------
__global__ void k_scan_chain(const float* __restrict__ S, const float* __restrict__ decay,
                             float* __restrict__ Hinit) {
  int gid = blockIdx.x * blockDim.x + threadIdx.x;
  if (gid >= BB * NH * HD * DST) return;
  int n = gid & (DST - 1);
  int p = (gid >> 7) % HD;
  int h = (gid / (DST * HD)) & (NH - 1);
  int b = gid / (DST * HD * NH);
  float hp = 0.f;
  for (int cc = 0; cc < NCHK; cc++) {
    size_t idx = ((((size_t)cc * BB + b) * NH + h) * HD + p) * DST + n;
    Hinit[idx] = hp;
    hp = hp * decay[(cc * BB + b) * NH + h] + S[idx];
  }
}

// ---------------- scan pass 3: per-chunk outputs with chained init ----------------
__global__ __launch_bounds__(128)
void k_scan_out(const float* __restrict__ xbc, const float* __restrict__ dtp,
                const float* __restrict__ dAv, const float* __restrict__ Hinit,
                const float* __restrict__ Dp, float* __restrict__ y) {
  const int c = blockIdx.x, b = blockIdx.y, h = blockIdx.z;
  const int tid = threadIdx.x;
  __shared__ float Bsh[DST], Csh[DST];
  __shared__ float sc2[2];
  float hreg[DST];
  const float Dh = Dp[h];
  if (tid < HD) {
    size_t sbase = ((((size_t)c * BB + b) * NH + h) * HD + tid) * DST;
#pragma unroll
    for (int n = 0; n < DST; n++) hreg[n] = Hinit[sbase + n];
  }

  for (int t = 0; t < LC; t++) {
    int row = b * LL + c * LC + t;
    Bsh[tid] = xbc[(size_t)row * CCH + DIN + tid];
    Csh[tid] = xbc[(size_t)row * CCH + DIN + DST + tid];
    if (tid == 0) { sc2[0] = dtp[row * NH + h]; sc2[1] = dAv[row * NH + h]; }
    __syncthreads();
    float dt = sc2[0], dA = sc2[1];
    if (tid < HD) {
      float xv  = xbc[(size_t)row * CCH + h * HD + tid];
      float dtx = dt * xv;
      float acc = 0.f;
#pragma unroll
      for (int n = 0; n < DST; n++) {
        hreg[n] = hreg[n] * dA + dtx * Bsh[n];
        acc += hreg[n] * Csh[n];
      }
      y[(size_t)row * DIN + h * HD + tid] = acc + Dh * xv;
    }
    __syncthreads();
  }
}

// ---------------- gated RMSNorm -> bf16 ----------------
__global__ __launch_bounds__(256)
void k_gate_norm(const float* __restrict__ zx, const float* __restrict__ y,
                 const float* __restrict__ norm_w, unsigned short* __restrict__ g) {
  const int row = blockIdx.x;
  const int tid = threadIdx.x;
  __shared__ float red[256];
  float gv[5];
  float ss = 0.f;
#pragma unroll
  for (int i = 0; i < 5; i++) {
    int cidx = tid + 256 * i;                       // 1280 = 5*256
    float z = zx[(size_t)row * DPROJ + cidx];
    float s = z / (1.0f + __expf(-z));              // SiLU(z)
    float v = y[(size_t)row * DIN + cidx] * s;
    gv[i] = v;
    ss += v * v;
  }
  red[tid] = ss;
  __syncthreads();
  for (int s = 128; s > 0; s >>= 1) {
    if (tid < s) red[tid] += red[tid + s];
    __syncthreads();
  }
  float r = rsqrtf(red[0] / (float)DIN + 1e-5f);
#pragma unroll
  for (int i = 0; i < 5; i++) {
    int cidx = tid + 256 * i;
    g[(size_t)row * DIN + cidx] = f32_to_bf16(gv[i] * r * norm_w[cidx]);
  }
}

// ---------------- host launcher ----------------
extern "C" void kernel_launch(void* const* d_in, const int* in_sizes, int n_in,
                              void* d_out, int out_size, void* d_ws, size_t ws_size,
                              hipStream_t stream) {
  const float* x       = (const float*)d_in[0];
  const float* W_in    = (const float*)d_in[1];
  const float* conv_w  = (const float*)d_in[2];
  const float* conv_b  = (const float*)d_in[3];
  const float* dt_bias = (const float*)d_in[4];
  const float* A_log   = (const float*)d_in[5];
  const float* Dp      = (const float*)d_in[6];
  const float* norm_w  = (const float*)d_in[7];
  const float* W_out   = (const float*)d_in[8];
  float* out = (float*)d_out;

  char* ws = (char*)d_ws;
  unsigned short* xbf   = (unsigned short*)(ws + OFF_XBF);
  unsigned short* winT  = (unsigned short*)(ws + OFF_WINT);
  float*          zx    = (float*)(ws + OFF_ZX);
  float*          xbc   = (float*)(ws + OFF_XBC);
  float*          dtp   = (float*)(ws + OFF_DTP);
  float*          dAv   = (float*)(ws + OFF_DAV);
  float*          S     = (float*)(ws + OFF_S);
  float*          Hinit = (float*)(ws + OFF_HINIT);
  float*          decay = (float*)(ws + OFF_DECAY);
  float*          yv    = (float*)(ws + OFF_Y);
  unsigned short* g     = (unsigned short*)(ws + OFF_G);
  unsigned short* woutT = (unsigned short*)(ws + OFF_WOUTT);

  // 1) casts / transposes for tensor-core GEMMs
  {
    int n = BL * DM;
    k_cast_bf16<<<(n + 255) / 256, 256, 0, stream>>>(x, xbf, n);
  }
  {
    int n = DPROJ * DM;
    k_transpose_cast_bf16<<<(n + 255) / 256, 256, 0, stream>>>(W_in, winT, DM, DPROJ);
  }
  {
    int n = DIN * DM;
    k_transpose_cast_bf16<<<(n + 255) / 256, 256, 0, stream>>>(W_out, woutT, DIN, DM);
  }

  // 2) in-projection GEMM: zx[8192 x 2832] = x[8192 x 1280] @ W_in
  {
    dim3 grid(BL / 128, (DPROJ + 127) / 128);
    k_gemm_bf16<<<grid, 256, 0, stream>>>(xbf, winT, zx, BL, DPROJ, DM);
  }

  // 3) conv + SiLU ; dt preprocessing
  {
    int n = BL * CCH;
    k_conv_silu<<<(n + 255) / 256, 256, 0, stream>>>(zx, conv_w, conv_b, xbc);
  }
  {
    int n = BL * NH;
    k_dt<<<(n + 255) / 256, 256, 0, stream>>>(zx, dt_bias, A_log, dtp, dAv);
  }

  // 4) chunked selective scan
  {
    dim3 grid(NCHK, BB, NH);
    k_scan_state<<<grid, 128, 0, stream>>>(xbc, dtp, dAv, A_log, S, decay);
  }
  {
    int n = BB * NH * HD * DST;
    k_scan_chain<<<(n + 255) / 256, 256, 0, stream>>>(S, decay, Hinit);
  }
  {
    dim3 grid(NCHK, BB, NH);
    k_scan_out<<<grid, 128, 0, stream>>>(xbc, dtp, dAv, Hinit, Dp, yv);
  }

  // 5) gated RMSNorm -> bf16
  k_gate_norm<<<BL, 256, 0, stream>>>(zx, yv, norm_w, g);

  // 6) out-projection GEMM: out[8192 x 1280] = g @ W_out
  {
    dim3 grid(BL / 128, DM / 128);
    k_gemm_bf16<<<grid, 256, 0, stream>>>(g, woutT, out, BL, DM, DIN);
  }
}
